// MoE_13967233646800
// MI455X (gfx1250) — matrix operand order repaired and checked by reading, not compiled, not run on hardware
//
#include <hip/hip_runtime.h>
#include <hip/hip_bf16.h>

// ---------------- problem constants (match reference) ----------------
#define T_TOK 4096
#define DDIM  1024
#define HDIM  4096
#define NEXP  8
#define TOPK  2

typedef __attribute__((ext_vector_type(16))) _Float16 v16h;
typedef __attribute__((ext_vector_type(8)))  float    v8f;

#define LDA 34   // LDS pitch in halves (17 dwords -> bank-conflict-friendly)
#define LDB 34

#if __has_builtin(__builtin_amdgcn_global_load_async_to_lds_b32)
#define HAVE_ASYNC_LDS 1
typedef __attribute__((address_space(1))) int* gptr_t;
typedef __attribute__((address_space(3))) int* lptr_t;
#else
#define HAVE_ASYNC_LDS 0
#endif
#if __has_builtin(__builtin_amdgcn_s_wait_asynccnt)
#define WAIT_ASYNC() __builtin_amdgcn_s_wait_asynccnt(0)
#else
#define WAIT_ASYNC() ((void)0)
#endif

// =====================================================================
// 0) zero per-expert counters (ws is only poisoned once; do this every call)
// =====================================================================
__global__ void moe_init(int* counts) {
    if (threadIdx.x < NEXP) counts[threadIdx.x] = 0;
}

// =====================================================================
// 1) gating: one wave (32 lanes) per token
// =====================================================================
__global__ __launch_bounds__(128) void moe_gate(
    const float* __restrict__ x, const float* __restrict__ Wg,
    const float* __restrict__ bg,
    int* counts, int* lists, int* eid, int* slotm, float* wgt)
{
    const int lane = threadIdx.x & 31;
    const int wib  = threadIdx.x >> 5;
    const int t    = blockIdx.x * 4 + wib;
    if (t >= T_TOK) return;

    float acc[NEXP];
#pragma unroll
    for (int e = 0; e < NEXP; ++e) acc[e] = 0.f;

    const float* xr = x + (size_t)t * DDIM;
    for (int d = lane; d < DDIM; d += 32) {
        const float xv = xr[d];
        const float* wr = Wg + (size_t)d * NEXP;
#pragma unroll
        for (int e = 0; e < NEXP; ++e) acc[e] += xv * wr[e];
    }
#pragma unroll
    for (int e = 0; e < NEXP; ++e) {
#pragma unroll
        for (int off = 16; off > 0; off >>= 1)
            acc[e] += __shfl_xor(acc[e], off, 32);
    }

    if (lane == 0) {
        float lg[NEXP], p[NEXP];
        float mx = -1e30f;
#pragma unroll
        for (int e = 0; e < NEXP; ++e) { lg[e] = acc[e] + bg[e]; mx = fmaxf(mx, lg[e]); }
        float se = 0.f;
#pragma unroll
        for (int e = 0; e < NEXP; ++e) { p[e] = expf(lg[e] - mx); se += p[e]; }
        const float inv = 1.0f / se;
#pragma unroll
        for (int e = 0; e < NEXP; ++e) p[e] *= inv;

        // top-2, lowest-index tie-break (matches jax.lax.top_k)
        int e0 = 0;
#pragma unroll
        for (int e = 1; e < NEXP; ++e) if (p[e] > p[e0]) e0 = e;
        int e1 = (e0 == 0) ? 1 : 0;
#pragma unroll
        for (int e = 0; e < NEXP; ++e) if (e != e0 && p[e] > p[e1]) e1 = e;

        const int es[2] = { e0, e1 };
#pragma unroll
        for (int k = 0; k < TOPK; ++k) {
            const int e    = es[k];
            const int slot = atomicAdd(&counts[e], 1);
            lists[(size_t)e * T_TOK + slot] = t;
            eid  [2 * t + k] = e;
            slotm[2 * t + k] = slot;
            wgt  [2 * t + k] = p[e];
        }
    }
}

// =====================================================================
// 2) tiny exclusive scan of 8 counts
// =====================================================================
__global__ void moe_scan(const int* counts, int* offsets) {
    if (threadIdx.x == 0) {
        int s = 0;
#pragma unroll
        for (int e = 0; e < NEXP; ++e) { offsets[e] = s; s += counts[e]; }
        offsets[NEXP] = s;
    }
}

// =====================================================================
// shared WMMA tile helpers (layouts per CDNA5 ISA 7.12.2)
// =====================================================================
__device__ __forceinline__ v16h load_a_frag(const _Float16* Al, int row, int lane) {
    const int ml = lane & 15;
    const int kb = (lane < 16) ? 0 : 8;          // 16-bit A 16x32 layout
    const _Float16* ap = Al + (size_t)(row + ml) * LDA;
    v16h a;
#pragma unroll
    for (int j = 0; j < 8; ++j) a[j]     = ap[kb + j];
#pragma unroll
    for (int j = 0; j < 8; ++j) a[8 + j] = ap[kb + 16 + j];
    return a;
}
__device__ __forceinline__ v16h load_b_frag(const _Float16* Bl, int col, int lane) {
    const int nl = lane & 15;
    const int kb = (lane < 16) ? 0 : 8;          // 16-bit B 32x16 layout
    const _Float16* bp = Bl + (size_t)(col + nl) * LDB + kb;
    v16h b;
#pragma unroll
    for (int j = 0; j < 8; ++j) b[j] = bp[j];
#pragma unroll
    for (int j = 0; j < 8; ++j) b[8 + j] = bp[16 + j];
    return b;
}

// =====================================================================
// 3) GEMM1: h = relu(gather(x) @ W1[e] + b1[e])  -> f16 hbuf
//    block tile 128(M) x 256(N), 8 waves as 2(M) x 4(N), 64x64 per wave
// =====================================================================
__global__ __launch_bounds__(256) void moe_gemm1(
    const float* __restrict__ x, const float* __restrict__ W1,
    const float* __restrict__ b1,
    const int* counts, const int* offsets, const int* lists,
    _Float16* __restrict__ hbuf)
{
    const int e  = blockIdx.z;
    const int Ne = counts[e];
    const int m0 = blockIdx.x * 128;
    if (m0 >= Ne) return;
    const int n0  = blockIdx.y * 256;
    const int off = offsets[e];
    const float* W = W1 + (size_t)e * DDIM * HDIM;

    __shared__ _Float16 Al[128 * LDA];
    __shared__ _Float16 Bl[256 * LDB];
    __shared__ int toks[128];

    const int tid = threadIdx.x, lane = tid & 31, wid = tid >> 5;
    const int wrow0 = (wid >> 2) * 64;  // 2 waves along M
    const int wcol0 = (wid & 3) * 64;   // 4 waves along N

    if (tid < 128) { const int m = m0 + tid; toks[tid] = (m < Ne) ? lists[(size_t)e * T_TOK + m] : -1; }
    __syncthreads();

    const v8f z8 = {0, 0, 0, 0, 0, 0, 0, 0};
    v8f acc[4][4];
#pragma unroll
    for (int i = 0; i < 4; ++i)
#pragma unroll
        for (int j = 0; j < 4; ++j) acc[i][j] = z8;

    for (int kk = 0; kk < DDIM; kk += 32) {
        // ---- stage A: gathered x rows, fp32 -> f16 (128 x 32)
        {
            const int r  = tid >> 1;
            const int cb = (tid & 1) * 16;
            const int tok = toks[r];
            _Float16* dst = &Al[(size_t)r * LDA + cb];
            if (tok >= 0) {
                const float* xr = x + (size_t)tok * DDIM + kk + cb;
#pragma unroll
                for (int j = 0; j < 4; ++j) {
                    const float4 v = *(const float4*)(xr + 4 * j);
                    dst[4*j+0] = (_Float16)v.x; dst[4*j+1] = (_Float16)v.y;
                    dst[4*j+2] = (_Float16)v.z; dst[4*j+3] = (_Float16)v.w;
                }
            } else {
#pragma unroll
                for (int j = 0; j < 16; ++j) dst[j] = (_Float16)0.0f;
            }
        }
        // ---- stage B transposed: Bl[n][k], one column per thread (32 x 256)
        {
            const int c = tid;                    // 0..255
            const float* Wp = W + (size_t)kk * HDIM + n0 + c;
#pragma unroll
            for (int kr = 0; kr < 32; ++kr)
                Bl[(size_t)c * LDB + kr] = (_Float16)Wp[(size_t)kr * HDIM];
            if (kk + 32 < DDIM)
                __builtin_prefetch(Wp + (size_t)32 * HDIM, 0, 1);  // global_prefetch_b8
        }
        __syncthreads();

        v16h af[4], bf[4];
#pragma unroll
        for (int ms = 0; ms < 4; ++ms) af[ms] = load_a_frag(Al, wrow0 + ms * 16, lane);
#pragma unroll
        for (int ns = 0; ns < 4; ++ns) bf[ns] = load_b_frag(Bl, wcol0 + ns * 16, lane);
#pragma unroll
        for (int ms = 0; ms < 4; ++ms)
#pragma unroll
            for (int ns = 0; ns < 4; ++ns)
                acc[ms][ns] = __builtin_amdgcn_wmma_f32_16x16x32_f16(
                    false, af[ms], false, bf[ns], (short)0, acc[ms][ns], false, false);
        __syncthreads();
    }

    // epilogue: +bias, relu, f16 store (C layout: M = r + 8*(lane>=16), N = lane&15)
    const int ml8 = (lane >> 4) * 8;
    const int nl  = lane & 15;
#pragma unroll
    for (int ms = 0; ms < 4; ++ms)
#pragma unroll
        for (int ns = 0; ns < 4; ++ns)
#pragma unroll
            for (int r = 0; r < 8; ++r) {
                const int m = m0 + wrow0 + ms * 16 + ml8 + r;
                if (m < Ne) {
                    const int gc = n0 + wcol0 + ns * 16 + nl;
                    float v = acc[ms][ns][r] + b1[(size_t)e * HDIM + gc];
                    v = fmaxf(v, 0.0f);
                    hbuf[(size_t)(off + m) * HDIM + gc] = (_Float16)v;
                }
            }
}

// =====================================================================
// 4) GEMM2: y = h @ W2[e] + b2[e]  -> f32 ybuf   (K over H=4096)
//    A tile is already f16 -> stage via async global->LDS when available
// =====================================================================
__global__ __launch_bounds__(256) void moe_gemm2(
    const _Float16* __restrict__ hbuf, const float* __restrict__ W2,
    const float* __restrict__ b2,
    const int* counts, const int* offsets,
    float* __restrict__ ybuf)
{
    const int e  = blockIdx.z;
    const int Ne = counts[e];
    const int m0 = blockIdx.x * 128;
    if (m0 >= Ne) return;
    const int n0  = blockIdx.y * 256;
    const int off = offsets[e];
    const float* W = W2 + (size_t)e * HDIM * DDIM;

    __shared__ _Float16 Al[128 * LDA];
    __shared__ _Float16 Bl[256 * LDB];

    const int tid = threadIdx.x, lane = tid & 31, wid = tid >> 5;
    const int wrow0 = (wid >> 2) * 64;
    const int wcol0 = (wid & 3) * 64;

    const v8f z8 = {0, 0, 0, 0, 0, 0, 0, 0};
    v8f acc[4][4];
#pragma unroll
    for (int i = 0; i < 4; ++i)
#pragma unroll
        for (int j = 0; j < 4; ++j) acc[i][j] = z8;

    for (int kk = 0; kk < HDIM; kk += 32) {
        // ---- stage A: f16 rows of hbuf (128 x 32 halves)
        {
            const int r  = tid >> 1;
            const int cb = (tid & 1) * 16;
            const int m  = m0 + r;
            unsigned* dst = (unsigned*)&Al[(size_t)r * LDA + cb];
            if (m < Ne) {
                const unsigned* src =
                    (const unsigned*)(hbuf + (size_t)(off + m) * HDIM + kk + cb);
#if HAVE_ASYNC_LDS
                // CDNA5 async global->LDS copy (ASYNCcnt-tracked, no VGPR bounce)
#pragma unroll
                for (int j = 0; j < 8; ++j)
                    __builtin_amdgcn_global_load_async_to_lds_b32(
                        (gptr_t)(src + j), (lptr_t)(dst + j), 0, 0);
#else
#pragma unroll
                for (int j = 0; j < 8; ++j) dst[j] = src[j];
#endif
            } else {
#pragma unroll
                for (int j = 0; j < 8; ++j) dst[j] = 0u;
            }
        }
        // ---- stage B transposed: Bl[n][k], one column per thread (32 x 256)
        {
            const int c = tid;
            const float* Wp = W + (size_t)kk * DDIM + n0 + c;
#pragma unroll
            for (int kr = 0; kr < 32; ++kr)
                Bl[(size_t)c * LDB + kr] = (_Float16)Wp[(size_t)kr * DDIM];
            if (kk + 32 < HDIM)
                __builtin_prefetch(Wp + (size_t)32 * DDIM, 0, 1);
        }
        WAIT_ASYNC();          // drain this wave's async LDS writes
        __syncthreads();

        v16h af[4], bf[4];
#pragma unroll
        for (int ms = 0; ms < 4; ++ms) af[ms] = load_a_frag(Al, wrow0 + ms * 16, lane);
#pragma unroll
        for (int ns = 0; ns < 4; ++ns) bf[ns] = load_b_frag(Bl, wcol0 + ns * 16, lane);
#pragma unroll
        for (int ms = 0; ms < 4; ++ms)
#pragma unroll
            for (int ns = 0; ns < 4; ++ns)
                acc[ms][ns] = __builtin_amdgcn_wmma_f32_16x16x32_f16(
                    false, af[ms], false, bf[ns], (short)0, acc[ms][ns], false, false);
        __syncthreads();
    }

    const int ml8 = (lane >> 4) * 8;
    const int nl  = lane & 15;
#pragma unroll
    for (int ms = 0; ms < 4; ++ms)
#pragma unroll
        for (int ns = 0; ns < 4; ++ns)
#pragma unroll
            for (int r = 0; r < 8; ++r) {
                const int m = m0 + wrow0 + ms * 16 + ml8 + r;
                if (m < Ne) {
                    const int gc = n0 + wcol0 + ns * 16 + nl;
                    ybuf[(size_t)(off + m) * DDIM + gc] =
                        acc[ms][ns][r] + b2[(size_t)e * DDIM + gc];
                }
            }
}

// =====================================================================
// 5) combine: out[t] = w0*y[row0] + w1*y[row1]  (fixed order -> deterministic)
// =====================================================================
__global__ __launch_bounds__(256) void moe_combine(
    const float* __restrict__ ybuf, const int* offsets,
    const int* eid, const int* slotm, const float* wgt,
    float* __restrict__ out)
{
    const int t = blockIdx.x;
    const int d = threadIdx.x * 4;
    float4 r = {0.f, 0.f, 0.f, 0.f};
#pragma unroll
    for (int k = 0; k < TOPK; ++k) {
        const int   e   = eid[2 * t + k];
        const int   row = offsets[e] + slotm[2 * t + k];
        const float w   = wgt[2 * t + k];
        const float4 y  = *(const float4*)(ybuf + (size_t)row * DDIM + d);
        r.x += w * y.x; r.y += w * y.y; r.z += w * y.z; r.w += w * y.w;
    }
    *(float4*)(out + (size_t)t * DDIM + d) = r;
}

// =====================================================================
// launcher
// =====================================================================
extern "C" void kernel_launch(void* const* d_in, const int* in_sizes, int n_in,
                              void* d_out, int out_size, void* d_ws, size_t ws_size,
                              hipStream_t stream) {
    const float* x  = (const float*)d_in[0];
    const float* Wg = (const float*)d_in[1];
    const float* bg = (const float*)d_in[2];
    const float* W1 = (const float*)d_in[3];
    const float* b1 = (const float*)d_in[4];
    const float* W2 = (const float*)d_in[5];
    const float* b2 = (const float*)d_in[6];
    float* out = (float*)d_out;

    // workspace carve-out (256B-aligned slices)
    char* w = (char*)d_ws;
    auto carve = [&](size_t bytes) -> void* {
        void* p = (void*)w;
        w += (bytes + 255) & ~(size_t)255;
        return p;
    };
    int*      counts  = (int*)carve(NEXP * sizeof(int));
    int*      offsets = (int*)carve((NEXP + 1) * sizeof(int));
    int*      lists   = (int*)carve((size_t)NEXP * T_TOK * sizeof(int));
    int*      eid     = (int*)carve((size_t)T_TOK * TOPK * sizeof(int));
    int*      slotm   = (int*)carve((size_t)T_TOK * TOPK * sizeof(int));
    float*    wgt     = (float*)carve((size_t)T_TOK * TOPK * sizeof(float));
    _Float16* hbuf    = (_Float16*)carve((size_t)TOPK * T_TOK * HDIM * sizeof(_Float16));
    float*    ybuf    = (float*)carve((size_t)TOPK * T_TOK * DDIM * sizeof(float));

    moe_init<<<1, 32, 0, stream>>>(counts);
    moe_gate<<<T_TOK / 4, 128, 0, stream>>>(x, Wg, bg, counts, lists, eid, slotm, wgt);
    moe_scan<<<1, 32, 0, stream>>>(counts, offsets);

    dim3 g1((T_TOK + 127) / 128, HDIM / 256, NEXP);   // worst-case M grid; blocks self-prune
    moe_gemm1<<<g1, 256, 0, stream>>>(x, W1, b1, counts, offsets, lists, hbuf);

    dim3 g2((T_TOK + 127) / 128, DDIM / 256, NEXP);
    moe_gemm2<<<g2, 256, 0, stream>>>(hbuf, W2, b2, counts, offsets, ybuf);

    moe_combine<<<T_TOK, DDIM / 4, 0, stream>>>(ybuf, offsets, eid, slotm, wgt, out);
}
